// ModulatedConv2d_13400297964131
// MI455X (gfx1250) — matrix operand order repaired and checked by reading, not compiled, not run on hardware
//
#include <hip/hip_runtime.h>

// ---------------------------------------------------------------------------
// ModulatedConv2d (StyleGAN2), b=16, ic=oc=512, k=3, h=w=64, sd=512.
// out[b,oc] = demod[b,oc] * conv(x[b,ic]*s[b,ic], w) + bias
// Implicit GEMM on v_wmma_f32_16x16x32_bf16 (wave32, gfx1250).
// Double-buffered LDS; A: register prefetch + pk cvt; B: async global->LDS.
// ---------------------------------------------------------------------------

#define B_   16
#define IC_  512
#define OC_  512
#define HW_  64
#define SD_  512
#define KTOT 4608   // ic * 9
#define PIX  4096   // 64*64
#define NK   144    // K-steps of 32
#define LDSTRIDE 40 // halfs per LDS row (80B: 16B aligned, 20-dword bank stride)

#define AS1 __attribute__((address_space(1)))
#define AS3 __attribute__((address_space(3)))

typedef __attribute__((__vector_size__(16))) int vint4;   // native int4 vector

#if defined(__has_builtin)
#  if __has_builtin(__builtin_amdgcn_global_load_async_to_lds_b128)
#    define HAVE_ASYNC_LDS 1
#  endif
#  if __has_builtin(__builtin_amdgcn_s_wait_asynccnt)
#    define HAVE_WAIT_ASYNC 1
#  endif
#endif

#if defined(HAVE_ASYNC_LDS)
#  if defined(HAVE_WAIT_ASYNC)
#    define WAIT_ASYNC() __builtin_amdgcn_s_wait_asynccnt(0)
#  else
#    define WAIT_ASYNC() asm volatile("s_wait_asynccnt 0x0" ::: "memory")
#  endif
#  define ASYNC_B128(gsrc, ldst) \
     __builtin_amdgcn_global_load_async_to_lds_b128( \
         (AS1 vint4*)(gsrc), (AS3 vint4*)(ldst), 0, 0)
#endif

typedef __attribute__((ext_vector_type(16))) __bf16 v16bf;
typedef __attribute__((ext_vector_type(8)))  float  v8f;

union FragAB { uint4 u[2]; v16bf v; };
union Pack16 { __bf16 h[16]; uint4 q[2]; };

__device__ __forceinline__ unsigned short f2bf(float f) {
    union { float f; unsigned int u; } cv; cv.f = f;
    unsigned int u = cv.u;
    unsigned int r = u + 0x7FFFu + ((u >> 16) & 1u);
    return (unsigned short)(r >> 16);
}

// ---- prep 1: s[b,ic] = style[b,:] . mod_w[ic,:] + mod_b[ic] ----------------
__global__ __launch_bounds__(256) void style_mod_kernel(
    const float* __restrict__ style, const float* __restrict__ mod_w,
    const float* __restrict__ mod_b, float* __restrict__ s)
{
    int idx = blockIdx.x * 256 + threadIdx.x;
    int b = idx / IC_, ic = idx % IC_;
    float acc = mod_b[ic];
    const float* st = style + b * SD_;
    const float* mw = mod_w + ic * SD_;
    for (int j = 0; j < SD_; ++j) acc += st[j] * mw[j];
    s[idx] = acc;
}

// ---- prep 2: wsq[oc,ic] = sum_k w[oc,ic,k]^2 -------------------------------
__global__ __launch_bounds__(256) void wsq_kernel(
    const float* __restrict__ w, float* __restrict__ wsq)
{
    int idx = blockIdx.x * 256 + threadIdx.x;
    const float* p = w + (size_t)idx * 9;
    float acc = 0.f;
#pragma unroll
    for (int k = 0; k < 9; ++k) { float v = p[k]; acc += v * v; }
    wsq[idx] = acc;
}

// ---- prep 3: demod[b,oc] = rsqrt(sum_ic s^2 * wsq + eps) -------------------
__global__ __launch_bounds__(256) void demod_kernel(
    const float* __restrict__ s, const float* __restrict__ wsq,
    float* __restrict__ demod)
{
    int idx = blockIdx.x * 256 + threadIdx.x;
    int b = idx / OC_, oc = idx % OC_;
    float acc = 1e-8f;
    const float* sp = s + b * IC_;
    const float* wp = wsq + oc * IC_;
    for (int ic = 0; ic < IC_; ++ic) { float sv = sp[ic]; acc += sv * sv * wp[ic]; }
    demod[idx] = rsqrtf(acc);
}

// ---- prep 4: repack weight f32 [oc][ic][3][3] -> bf16 [oc][tap*512+ic] -----
__global__ __launch_bounds__(256) void wconv_kernel(
    const float* __restrict__ w, unsigned short* __restrict__ wb)
{
    int idx = blockIdx.x * 256 + threadIdx.x;
    int oc = idx / KTOT;
    int r  = idx % KTOT;
    int tap = r / IC_, ic = r % IC_;
    wb[idx] = f2bf(w[((size_t)oc * IC_ + ic) * 9 + tap]);
}

// ---- main: implicit-GEMM conv on WMMA bf16 ---------------------------------
// grid: (32 pixel-tiles, 4 oc-tiles, 16 batch), block 256 (8 waves).
// Block tile: M=128 pixels (2 image rows) x N=128 oc, K-step 32.
// Wave tile: 32 pix x 64 oc => 2x4 WMMA 16x16 accumulators.
__global__ __launch_bounds__(256) void conv_kernel(
    const float* __restrict__ x, const float* __restrict__ s,
    const float* __restrict__ demod, const unsigned short* __restrict__ wb,
    const float* __restrict__ bias, float* __restrict__ out)
{
    __shared__ unsigned short sA[2][128 * LDSTRIDE];  // im2col tile [pix][k]
    __shared__ unsigned short sB[2][128 * LDSTRIDE];  // weight tile [oc][k]
    __shared__ float sSh[IC_];                        // per-sample modulation

    const int tid  = threadIdx.x;
    const int b    = blockIdx.z;
    const int ocb  = blockIdx.y * 128;
    const int y0   = blockIdx.x * 2;
    const int lane = tid & 31;
    const int wave = tid >> 5;
    const int wm   = wave & 3;
    const int wn   = wave >> 2;

    // stage style modulation once
    sSh[tid]       = s[b * IC_ + tid];
    sSh[tid + 256] = s[b * IC_ + 256 + tid];
    __syncthreads();

    v8f c[2][4];
#pragma unroll
    for (int mi = 0; mi < 2; ++mi)
#pragma unroll
        for (int ni = 0; ni < 4; ++ni) c[mi][ni] = (v8f){};

    // A loader: 128 threads cover the 128 pixels; licg picks 16 contiguous ic
    const int lx     = tid & 63;
    const int lyrow  = (tid >> 6) & 1;
    const int licg   = tid >> 7;                  // 0/1 -> ic 0..15 / 16..31
    const int plocal = lyrow * 64 + lx;
    const int aStOff = plocal * LDSTRIDE + licg * 16;   // halfs
    // B loader: 2 threads per oc row, 16 halfs each
    const int brow   = tid >> 1;
    const int bseg   = tid & 1;
    const int bStOff = brow * LDSTRIDE + bseg * 16;

    const float* xb = x + (size_t)b * IC_ * PIX;
    const int mrow  = lane & 15;
    const int khalf = lane >> 4;
    const int aRow0 = (wm * 32 + mrow) * LDSTRIDE;
    const int bRow0 = (wn * 64 + mrow) * LDSTRIDE;

    // ---- register prefetch state ----
    float ra[16];
    int   ra_sbase;                                // ic base for s-scale
    int   ndy = -1, ndx = -1, nic = 0;             // params of next K-step
    const unsigned short* wbp = wb + (size_t)(ocb + brow) * KTOT + bseg * 16;
#if !defined(HAVE_ASYNC_LDS)
    uint4 rb0, rb1;
#endif

    // ---- prefetch K-step 0 ----
    {
        const int yy = y0 + lyrow + ndy;
        const int xx = lx + ndx;
        const bool valid = ((unsigned)yy < HW_) && ((unsigned)xx < HW_);
        ra_sbase = nic * 32 + licg * 16;
        const float* src = xb + (size_t)ra_sbase * PIX + yy * HW_ + xx;
#pragma unroll
        for (int j = 0; j < 16; ++j) ra[j] = valid ? src[(size_t)j * PIX] : 0.f;
#if defined(HAVE_ASYNC_LDS)
        unsigned short* dB = sB[0] + bStOff;
        ASYNC_B128(wbp,     dB);
        ASYNC_B128(wbp + 8, dB + 8);
#else
        const uint4* bs = (const uint4*)wbp;
        rb0 = bs[0]; rb1 = bs[1];
#endif
        wbp += 32;
        if (++nic == 16) { nic = 0; if (++ndx == 2) { ndx = -1; ++ndy; } }
    }

    for (int kk = 0; kk < NK; ++kk) {
        unsigned short* bufA = sA[kk & 1];
        unsigned short* bufB = sB[kk & 1];

        // ---- store A tile (s-scale + bf16 pack), publish B tile ----
        {
            const float4* sp4 = (const float4*)(sSh + ra_sbase);
            float4 s0 = sp4[0], s1 = sp4[1], s2 = sp4[2], s3 = sp4[3];
            float sc[16] = { s0.x,s0.y,s0.z,s0.w, s1.x,s1.y,s1.z,s1.w,
                             s2.x,s2.y,s2.z,s2.w, s3.x,s3.y,s3.z,s3.w };
            Pack16 P;
#pragma unroll
            for (int j = 0; j < 16; ++j) P.h[j] = (__bf16)(ra[j] * sc[j]);
            uint4* dst = (uint4*)(bufA + aStOff);
            dst[0] = P.q[0]; dst[1] = P.q[1];
        }
#if defined(HAVE_ASYNC_LDS)
        WAIT_ASYNC();                  // our async B stores into bufB done
#else
        {
            uint4* dst = (uint4*)(bufB + bStOff);
            dst[0] = rb0; dst[1] = rb1;
        }
#endif
        __syncthreads();

        // ---- prefetch next K-step (overlaps with WMMA below) ----
        if (kk != NK - 1) {
            const int yy = y0 + lyrow + ndy;
            const int xx = lx + ndx;
            const bool valid = ((unsigned)yy < HW_) && ((unsigned)xx < HW_);
            ra_sbase = nic * 32 + licg * 16;
            const float* src = xb + (size_t)ra_sbase * PIX + yy * HW_ + xx;
#pragma unroll
            for (int j = 0; j < 16; ++j) ra[j] = valid ? src[(size_t)j * PIX] : 0.f;
#if defined(HAVE_ASYNC_LDS)
            unsigned short* dB = sB[(kk + 1) & 1] + bStOff;
            ASYNC_B128(wbp,     dB);
            ASYNC_B128(wbp + 8, dB + 8);
#else
            const uint4* bs = (const uint4*)wbp;
            rb0 = bs[0]; rb1 = bs[1];
#endif
            wbp += 32;
            if (++nic == 16) { nic = 0; if (++ndx == 2) { ndx = -1; ++ndy; } }
        }

        // ---- fragment loads (ISA 16-bit A 16x32 / B 32x16 lane layouts) ----
        FragAB a[2], bf4[4];
#pragma unroll
        for (int mi = 0; mi < 2; ++mi) {
            const uint4* p = (const uint4*)(bufA + aRow0 + mi * 16 * LDSTRIDE);
            a[mi].u[0] = p[khalf];
            a[mi].u[1] = p[2 + khalf];
        }
#pragma unroll
        for (int ni = 0; ni < 4; ++ni) {
            const uint4* p = (const uint4*)(bufB + bRow0 + ni * 16 * LDSTRIDE);
            bf4[ni].u[0] = p[khalf];
            bf4[ni].u[1] = p[2 + khalf];
        }
        // ---- 8 WMMAs ----
#pragma unroll
        for (int mi = 0; mi < 2; ++mi)
#pragma unroll
            for (int ni = 0; ni < 4; ++ni)
                c[mi][ni] = __builtin_amdgcn_wmma_f32_16x16x32_bf16(
                    false, a[mi].v, false, bf4[ni].v,
                    (short)0, c[mi][ni], false, false);
    }

    // ---- epilogue: acc * demod[b,oc] + bias[oc]; 2x float4 per fragment ----
#pragma unroll
    for (int ni = 0; ni < 4; ++ni) {
        const int oc = ocb + wn * 64 + ni * 16 + mrow;
        const float dm = demod[b * OC_ + oc];
        const float bi = bias[oc];
        float* obase = out + ((size_t)b * OC_ + oc) * PIX;
#pragma unroll
        for (int mi = 0; mi < 2; ++mi) {
            const int p0 = blockIdx.x * 128 + wm * 32 + mi * 16 + khalf * 8;
            v8f cc = c[mi][ni];
            float4 v0 = make_float4(cc[0]*dm+bi, cc[1]*dm+bi, cc[2]*dm+bi, cc[3]*dm+bi);
            float4 v1 = make_float4(cc[4]*dm+bi, cc[5]*dm+bi, cc[6]*dm+bi, cc[7]*dm+bi);
            float4* op = (float4*)(obase + p0);
            op[0] = v0; op[1] = v1;
        }
    }
}

// ---------------------------------------------------------------------------
extern "C" void kernel_launch(void* const* d_in, const int* in_sizes, int n_in,
                              void* d_out, int out_size, void* d_ws, size_t ws_size,
                              hipStream_t stream) {
    const float* x      = (const float*)d_in[0];
    const float* style  = (const float*)d_in[1];
    const float* weight = (const float*)d_in[2];
    const float* bias   = (const float*)d_in[3];
    const float* mod_w  = (const float*)d_in[4];
    const float* mod_b  = (const float*)d_in[5];
    float* out = (float*)d_out;

    char* ws = (char*)d_ws;
    float*          s_buf = (float*)(ws + 0);
    float*          dmod  = (float*)(ws + 32768);
    float*          wsq   = (float*)(ws + 65536);
    unsigned short* wbuf  = (unsigned short*)(ws + 65536 + 1048576);

    style_mod_kernel<<<(B_ * IC_) / 256, 256, 0, stream>>>(style, mod_w, mod_b, s_buf);
    wsq_kernel<<<(OC_ * IC_) / 256, 256, 0, stream>>>(weight, wsq);
    demod_kernel<<<(B_ * OC_) / 256, 256, 0, stream>>>(s_buf, wsq, dmod);
    wconv_kernel<<<(OC_ * KTOT) / 256, 256, 0, stream>>>(weight, wbuf);

    dim3 grid(PIX / 128, OC_ / 128, B_);
    conv_kernel<<<grid, 256, 0, stream>>>(x, s_buf, dmod, wbuf, bias, out);
}
